// ForwardGRUNet_74577812128046
// MI455X (gfx1250) — matrix operand order repaired and checked by reading, not compiled, not run on hardware
//
#include <hip/hip_runtime.h>

// ---------------------------------------------------------------------------
// 2-layer GRU LM forward for MI455X (gfx1250, wave32, WMMA bf16).
//   B=16, T=256, H=1024, V=32000. All GEMMs: bf16 inputs, f32 accumulate.
// This revision replaces global->VGPR->LDS staging with CDNA5 async
// global->LDS loads (ASYNCcnt) and double-buffers the GEMM LDS tiles.
// ---------------------------------------------------------------------------

typedef unsigned short u16;
typedef __attribute__((ext_vector_type(16))) __bf16        v16bf;
typedef __attribute__((ext_vector_type(8)))  float         v8f;
typedef __attribute__((ext_vector_type(8)))  unsigned short v8u16;
typedef __attribute__((ext_vector_type(16))) unsigned short v16u16;
typedef __attribute__((ext_vector_type(4)))  unsigned short v4u16;

#define FEAT  1024
#define G3    3072
#define BB    16
#define TT    256
#define MROWS 4096      // T*B, time-major rows
#define VOCABN 32000
#define NBLK_SCAN 64    // H/16 column slices

__device__ __forceinline__ u16 f2bf(float f) {
  unsigned u = __builtin_bit_cast(unsigned, f);
  u += 0x7FFFu + ((u >> 16) & 1u);        // round-to-nearest-even
  return (u16)(u >> 16);
}
__device__ __forceinline__ float bf2f(u16 h) {
  unsigned u = ((unsigned)h) << 16;
  return __builtin_bit_cast(float, u);
}

// CDNA5 async global->LDS copy, 16B per lane, tracked with ASYNCcnt.
// VDST = LDS byte offset (low 32 bits of a generic LDS pointer), VADDR = 64b
// global address, GV mode (saddr = off).
__device__ __forceinline__ void async_ld_b128(void* lds_dst, const void* gsrc) {
  unsigned l = (unsigned)(uintptr_t)lds_dst;
  unsigned long long g = (unsigned long long)(uintptr_t)gsrc;
  asm volatile("global_load_async_to_lds_b128 %0, %1, off"
               :: "v"(l), "v"(g) : "memory");
}
__device__ __forceinline__ void wait_async0() {
  asm volatile("s_wait_asynccnt 0x0" ::: "memory");
}

// A fragment (16x32 bf16, M=lane%16): lane<16 holds K0..15,
// lane>=16 holds K16..31; two contiguous 16B chunks.
__device__ __forceinline__ v16bf frag_a(const u16* row, int hl) {
  v8u16 c0 = *(const v8u16*)(row + 16 * hl);
  v8u16 c1 = *(const v8u16*)(row + 16 * hl + 8);
  v16u16 t;
#pragma unroll
  for (int i = 0; i < 8; ++i) { t[i] = c0[i]; t[i + 8] = c1[i]; }
  return __builtin_bit_cast(v16bf, t);
}
// B fragment (32x16 bf16, N=lane%16): lane<16 holds K0..15, lane>=16 K16..31;
// 32 contiguous bytes per lane from row-major [N][K] storage.
__device__ __forceinline__ v16bf frag_b(const u16* row, int hl) {
  v8u16 c0 = *(const v8u16*)(row + 16 * hl);
  v8u16 c1 = *(const v8u16*)(row + 16 * hl + 8);
  v16u16 t;
#pragma unroll
  for (int i = 0; i < 8; ++i) { t[i] = c0[i]; t[i + 8] = c1[i]; }
  return __builtin_bit_cast(v16bf, t);
}

// ---------------------------------------------------------------------------
// f32 -> bf16 bulk convert (n4 = number of float4 groups)
__global__ void conv_f32_bf16(const float* __restrict__ in,
                              u16* __restrict__ out, long n4) {
  long i = (long)blockIdx.x * blockDim.x + threadIdx.x;
  if (i >= n4) return;
  const float4 v = *(const float4*)(in + i * 4);
  v4u16 o; o[0] = f2bf(v.x); o[1] = f2bf(v.y); o[2] = f2bf(v.z); o[3] = f2bf(v.w);
  *(v4u16*)(out + i * 4) = o;
}

// Embedding gather into time-major bf16 rows (row = t*16 + b)
__global__ void embed_gather(const int* __restrict__ x,
                             const float* __restrict__ emb,
                             u16* __restrict__ xs) {
  int idx = blockIdx.x * blockDim.x + threadIdx.x;   // < 4096*256
  if (idx >= MROWS * 256) return;
  int row = idx >> 8;
  int c4  = (idx & 255) << 2;
  int t = row >> 4, b = row & 15;
  int tok = x[b * TT + t];
  const float4 v = *(const float4*)(emb + (size_t)tok * FEAT + c4);
  v4u16 o; o[0] = f2bf(v.x); o[1] = f2bf(v.y); o[2] = f2bf(v.z); o[3] = f2bf(v.w);
  *(v4u16*)(xs + (size_t)row * FEAT + c4) = o;
}

// ---------------------------------------------------------------------------
// Tiled bf16 WMMA GEMM: C[M,N] = A[M,K] * B[N,K]^T (+bias[N]).
// 256 threads = 8 waves (2x4), 128x128 block tile, wave tile 64x32.
// LDS tiles are double-buffered and filled with async global->LDS loads so
// the k+1 tile streams in underneath the 8 WMMAs of tile k.
// bt_remap: remap time-major row t*16+b -> output row b*256+t (decoder).
__global__ __launch_bounds__(256) void gemm_bf16(
    const u16* __restrict__ A, const u16* __restrict__ B,
    const float* __restrict__ bias, float* __restrict__ C,
    int K, int ldc, int bt_remap) {
  __shared__ u16 sA[2][128 * 40];   // 32-col K-chunk, stride 40 (bank padding)
  __shared__ u16 sB[2][128 * 40];

  const int tid  = threadIdx.x;
  const int wave = tid >> 5, lane = tid & 31;
  const int hl = lane >> 4, nn = lane & 15;
  const int wm = wave >> 2, wn = wave & 3;
  const int m0 = blockIdx.y * 128, n0 = blockIdx.x * 128;

  // per-thread chunk assignment (2 x 16B chunks of each tile)
  int cr[2], cc[2];
#pragma unroll
  for (int i = 0; i < 2; ++i) {
    int c = tid + i * 256;
    cr[i] = c >> 2;
    cc[i] = (c & 3) * 8;
  }

  auto issue = [&](int kk, int buf) {
#pragma unroll
    for (int i = 0; i < 2; ++i) {
      async_ld_b128(&sA[buf][cr[i] * 40 + cc[i]],
                    A + (size_t)(m0 + cr[i]) * K + kk + cc[i]);
      async_ld_b128(&sB[buf][cr[i] * 40 + cc[i]],
                    B + (size_t)(n0 + cr[i]) * K + kk + cc[i]);
    }
  };

  v8f acc[4][2] = {};
  const int KI = K / 32;

  issue(0, 0);
  for (int k = 0; k < KI; ++k) {
    wait_async0();        // own wave's async tile writes landed in LDS
    __syncthreads();      // every wave's tile writes landed; prev reads done
    if (k + 1 < KI) issue((k + 1) * 32, (k + 1) & 1);

    const u16* cA = sA[k & 1];
    const u16* cB = sB[k & 1];
    v16bf bf[2], af[4];
#pragma unroll
    for (int ns = 0; ns < 2; ++ns)
      bf[ns] = frag_b(cB + (wn * 32 + ns * 16 + nn) * 40, hl);
#pragma unroll
    for (int ms = 0; ms < 4; ++ms)
      af[ms] = frag_a(cA + (wm * 64 + ms * 16 + nn) * 40, hl);
#pragma unroll
    for (int ms = 0; ms < 4; ++ms)
#pragma unroll
      for (int ns = 0; ns < 2; ++ns)
        acc[ms][ns] = __builtin_amdgcn_wmma_f32_16x16x32_bf16(
            false, af[ms], false, bf[ns], (short)0, acc[ms][ns], false, false);
  }

#pragma unroll
  for (int ms = 0; ms < 4; ++ms) {
#pragma unroll
    for (int ns = 0; ns < 2; ++ns) {
      int col = n0 + wn * 32 + ns * 16 + nn;
      float bv = bias ? bias[col] : 0.f;
#pragma unroll
      for (int v = 0; v < 8; ++v) {
        int row  = m0 + wm * 64 + ms * 16 + v + 8 * hl;   // C layout: M=v+8*hl
        int orow = bt_remap ? ((row & 15) * TT + (row >> 4)) : row;
        C[(size_t)orow * ldc + col] = acc[ms][ns][v] + bv;
      }
    }
  }
}

// ---------------------------------------------------------------------------
// Persistent recurrent scan. 64 blocks x 96 threads (3 waves = gates r,z,n).
// Block owns H-columns [c0, c0+16). W_hh slice (3x16x1024 bf16) async-loaded
// to LDS once; h_prev async-staged to LDS per step; gate tiles exchanged via
// LDS; device-wide barrier between timesteps (monotonic atomic counter).
__global__ __launch_bounds__(96) void gru_scan(
    const float* __restrict__ xproj,   // [T*B, 3H] f32 (b_ih folded in)
    const u16*   __restrict__ Whh,     // [3H, H] bf16
    const float* __restrict__ bhh,     // [3H]
    u16*         __restrict__ hbuf,    // [2][B*H] bf16 ping-pong (slot0 init)
    u16*         __restrict__ ys,      // [T*B, H] bf16 layer output
    float*       __restrict__ hidden_out, // [B, H] final hidden (f32)
    unsigned*    __restrict__ cnt) {
  extern __shared__ u16 smem[];
  u16*  sW = smem;                         // 48 rows * 1032 (padded)
  u16*  sH = sW + 48 * 1032;               // 16 rows * 1032
  float* sG = (float*)(sH + 16 * 1032);    // r/z tiles: 2 * 256 f32

  const int tid = threadIdx.x;
  const int g = tid >> 5, lane = tid & 31;
  const int hl = lane >> 4, nn = lane & 15;
  const int c0 = blockIdx.x * 16;

  // Async-preload this block's W_hh slice (gate-major rows) into LDS, once.
  for (int c = tid; c < 48 * 128; c += 96) {
    int r = c >> 7, k8 = (c & 127) * 8;
    int gg = r >> 4, j = r & 15;
    async_ld_b128(sW + r * 1032 + k8,
                  Whh + (size_t)(gg * FEAT + c0 + j) * FEAT + k8);
  }

  const float bhh_v = bhh[g * FEAT + c0 + nn];

  for (int t = 0; t < TT; ++t) {
    const u16* hprev = hbuf + (t & 1) * (BB * FEAT);
    u16*       hnext = hbuf + ((t + 1) & 1) * (BB * FEAT);

    // async-stage h_prev (32KB bf16); first iteration also drains W preload
    for (int c = tid; c < 16 * 128; c += 96) {
      int r = c >> 7, k8 = (c & 127) * 8;
      async_ld_b128(sH + r * 1032 + k8, hprev + r * FEAT + k8);
    }
    wait_async0();
    __syncthreads();

    // hp[g] tile = h_prev @ W_hh[g]^T  (M=16 batch, N=16 cols, K=1024)
    v8f acc = {};
    for (int k0 = 0; k0 < FEAT; k0 += 32) {
      v16bf a = frag_a(sH + nn * 1032 + k0, hl);
      v16bf b = frag_b(sW + (g * 16 + nn) * 1032 + k0, hl);
      acc = __builtin_amdgcn_wmma_f32_16x16x32_bf16(
          false, a, false, b, (short)0, acc, false, false);
    }

    // matching x_proj values (b_ih already folded in by the GEMM)
    float xp[8];
#pragma unroll
    for (int v = 0; v < 8; ++v) {
      int b = v + 8 * hl;
      xp[v] = xproj[(size_t)(t * BB + b) * G3 + g * FEAT + c0 + nn];
    }

    if (g < 2) {                                   // r and z gates -> sigmoid
#pragma unroll
      for (int v = 0; v < 8; ++v) {
        float pre = acc[v] + bhh_v + xp[v];
        sG[g * 256 + v * 32 + lane] = 1.f / (1.f + __expf(-pre));
      }
    }
    __syncthreads();
    if (g == 2) {                                  // n gate + combine
#pragma unroll
      for (int v = 0; v < 8; ++v) {
        int b   = v + 8 * hl;
        float r = sG[v * 32 + lane];
        float z = sG[256 + v * 32 + lane];
        float n = tanhf(xp[v] + r * (acc[v] + bhh_v));
        float hp = bf2f(sH[b * 1032 + c0 + nn]);
        float hnew = (1.f - z) * n + z * hp;
        u16 hb = f2bf(hnew);
        hnext[b * FEAT + c0 + nn] = hb;
        ys[(size_t)(t * BB + b) * FEAT + c0 + nn] = hb;
        if (t == TT - 1) hidden_out[b * FEAT + c0 + nn] = hnew;
      }
    }

    // device-wide timestep barrier (monotonic counter, sleep-spin)
    __syncthreads();
    if (tid == 0) {
      __threadfence();
      atomicAdd(cnt, 1u);
      unsigned target = (unsigned)NBLK_SCAN * (unsigned)(t + 1);
      while (__hip_atomic_load(cnt, __ATOMIC_RELAXED, __HIP_MEMORY_SCOPE_AGENT) < target)
        __builtin_amdgcn_s_sleep(2);
    }
    __syncthreads();
    __threadfence();
  }
}

// ---------------------------------------------------------------------------
extern "C" void kernel_launch(void* const* d_in, const int* in_sizes, int n_in,
                              void* d_out, int out_size, void* d_ws, size_t ws_size,
                              hipStream_t stream) {
  const int*   x     = (const int*)d_in[0];
  const float* h0    = (const float*)d_in[1];
  const float* emb   = (const float*)d_in[2];
  const float* Wih0  = (const float*)d_in[3];
  const float* Whh0  = (const float*)d_in[4];
  const float* bih0  = (const float*)d_in[5];
  const float* bhh0  = (const float*)d_in[6];
  const float* Wih1  = (const float*)d_in[7];
  const float* Whh1  = (const float*)d_in[8];
  const float* bih1  = (const float*)d_in[9];
  const float* bhh1  = (const float*)d_in[10];
  const float* dec_b = (const float*)d_in[11];

  float* logits = (float*)d_out;                             // [B,T,V]
  float* hidden = (float*)d_out + (size_t)BB * TT * VOCABN;  // [2,B,H]

  // workspace carve-up (256B aligned)
  char* w = (char*)d_ws;
  size_t off = 0;
  auto carve = [&](size_t bytes) -> char* {
    char* p = w + off;
    off += (bytes + 255) & ~(size_t)255;
    return p;
  };
  u16*   emb_bf = (u16*)  carve((size_t)VOCABN * FEAT * 2);  // 65.5 MB
  u16*   xbufA  = (u16*)  carve((size_t)MROWS * FEAT * 2);   // xs, reused as ys0
  u16*   ys1    = (u16*)  carve((size_t)MROWS * FEAT * 2);
  float* xproj  = (float*)carve((size_t)MROWS * G3 * 4);     // 50.3 MB (reused)
  u16*   wib    = (u16*)  carve((size_t)G3 * FEAT * 2);      // W_ih bf16 (reused)
  u16*   whb    = (u16*)  carve((size_t)G3 * FEAT * 2);      // W_hh bf16 (reused)
  u16*   hbuf   = (u16*)  carve((size_t)2 * BB * FEAT * 2);
  unsigned* cnt = (unsigned*)carve(256);

  const int SCAN_SMEM = (48 * 1032 + 16 * 1032) * 2 + 512 * 4;  // ~131 KB
  hipFuncSetAttribute((const void*)gru_scan,
                      hipFuncAttributeMaxDynamicSharedMemorySize, SCAN_SMEM);

  hipMemsetAsync(cnt, 0, 256, stream);

  // emb -> bf16 ; embed gather
  {
    long n4 = (long)VOCABN * FEAT / 4;
    conv_f32_bf16<<<(unsigned)((n4 + 255) / 256), 256, 0, stream>>>(emb, emb_bf, n4);
    embed_gather<<<(MROWS * 256 + 255) / 256, 256, 0, stream>>>(x, emb, xbufA);
  }

  long n4w = (long)G3 * FEAT / 4;     // weight convert size
  long n4h = (long)BB * FEAT / 4;     // h0 slice convert size

  // ----- layer 0 -----
  conv_f32_bf16<<<(unsigned)((n4w + 255) / 256), 256, 0, stream>>>(Wih0, wib, n4w);
  gemm_bf16<<<dim3(G3 / 128, MROWS / 128), 256, 0, stream>>>(
      xbufA, wib, bih0, xproj, FEAT, G3, 0);
  conv_f32_bf16<<<(unsigned)((n4w + 255) / 256), 256, 0, stream>>>(Whh0, whb, n4w);
  conv_f32_bf16<<<(unsigned)((n4h + 255) / 256), 256, 0, stream>>>(h0, hbuf, n4h);
  gru_scan<<<NBLK_SCAN, 96, SCAN_SMEM, stream>>>(
      xproj, whb, bhh0, hbuf, /*ys0 reuses xs buffer*/ xbufA, hidden, cnt);

  // ----- layer 1 -----
  conv_f32_bf16<<<(unsigned)((n4w + 255) / 256), 256, 0, stream>>>(Wih1, wib, n4w);
  gemm_bf16<<<dim3(G3 / 128, MROWS / 128), 256, 0, stream>>>(
      xbufA, wib, bih1, xproj, FEAT, G3, 0);
  conv_f32_bf16<<<(unsigned)((n4w + 255) / 256), 256, 0, stream>>>(Whh1, whb, n4w);
  conv_f32_bf16<<<(unsigned)((n4h + 255) / 256), 256, 0, stream>>>(
      h0 + (size_t)BB * FEAT, hbuf, n4h);
  gru_scan<<<NBLK_SCAN, 96, SCAN_SMEM, stream>>>(
      xproj, whb, bhh1, hbuf, ys1, hidden + (size_t)BB * FEAT,
      (unsigned*)((char*)cnt + 128));

  // ----- tied decoder: logits = ys1 @ emb^T + dec_b -----
  gemm_bf16<<<dim3(VOCABN / 128, MROWS / 128), 256, 0, stream>>>(
      ys1, emb_bf, dec_b, logits, FEAT, VOCABN, 1);
}
